// MultiHeadAttention_25898652795249
// MI455X (gfx1250) — matrix operand order repaired and checked by reading, not compiled, not run on hardware
//
#include <hip/hip_runtime.h>

// ---------------- problem constants ----------------
constexpr int BS = 4, QLEN = 2048, DIM = 1024, NH = 16, DH = 64;
constexpr int NTOK = BS * QLEN;            // 8192 tokens
constexpr float QSCALE = 0.125f;           // 1/sqrt(DH)

// ---------------- WMMA helpers ----------------
typedef __attribute__((ext_vector_type(16))) __bf16 v16bf;
typedef __attribute__((ext_vector_type(8)))  float  v8f;

union FragBf {
  v16bf v;
  uint4 u[2];
};

__device__ __forceinline__ v8f wmma_bf16(const FragBf& a, const FragBf& b, v8f c) {
  // D = A(16x32 bf16) * B(32x16 bf16) + C(16x16 f32)
  return __builtin_amdgcn_wmma_f32_16x16x32_bf16(
      /*neg_a=*/false, a.v, /*neg_b=*/false, b.v,
      /*c_mod=*/(short)0, c, /*reuse_a=*/false, /*reuse_b=*/false);
}

// ---------------- Tensor Data Mover (TDM) helpers ----------------
typedef __attribute__((ext_vector_type(4))) unsigned int u32x4;
typedef __attribute__((ext_vector_type(8))) int          i32x8;
typedef __attribute__((ext_vector_type(4))) int          i32x4;

__device__ __forceinline__ unsigned lds_byte_off(const void* p) {
  // generic pointer to LDS: low 32 bits are the LDS byte offset
  return (unsigned)(uintptr_t)p;
}

// 2-D tile load: global (row-major, strideElems between rows) -> LDS,
// with TDM row padding (padIntervalCode/padAmountCode per D# encoding).
__device__ __forceinline__ void tdm_load_2d(
    unsigned ldsAddr, const void* gaddr,
    unsigned tensorD0, unsigned tensorD1,
    unsigned tileD0, unsigned tileD1,
    unsigned long long strideElems,
    int padIntervalCode, int padAmountCode) {
  union { unsigned long long q[2]; u32x4 v; } g0;
  unsigned long long ga = (unsigned long long)(uintptr_t)gaddr;
  g0.q[0] = 1ull /*count=1*/ | ((unsigned long long)ldsAddr << 32);
  g0.q[1] = (ga & 0x01FFFFFFFFFFFFFFull) | (2ull << 62);  // type=2 ("image")

  union { unsigned long long q[4]; i32x8 v; } g1;
  unsigned long long d0 = tensorD0, d1 = tensorD1, s0 = strideElems;
  unsigned long long cfg = (1ull << 16);                   // data_size = 2 bytes
  if (padAmountCode >= 0)
    cfg |= (1ull << 20) | ((unsigned long long)padIntervalCode << 22)
         | ((unsigned long long)padAmountCode << 25);
  g1.q[0] = cfg | ((d0 & 0xFFFFull) << 48);
  g1.q[1] = (d0 >> 16) | ((d1 & 0xFFFFFFFFull) << 16)
          | ((unsigned long long)(tileD0 & 0xFFFF) << 48);
  g1.q[2] = (unsigned long long)(tileD1 & 0xFFFF) | ((s0 & 0xFFFFFFFFull) << 32);
  g1.q[3] = (s0 >> 32) & 0xFFFFull;                        // tensor_dim1_stride = 0

  i32x4 z4 = {0, 0, 0, 0};
#if defined(__clang_major__) && (__clang_major__ >= 23)
  i32x8 z8 = {0, 0, 0, 0, 0, 0, 0, 0};
  __builtin_amdgcn_tensor_load_to_lds(g0.v, g1.v, z4, z4, z8, 0);
#else
  __builtin_amdgcn_tensor_load_to_lds(g0.v, g1.v, z4, z4, 0);
#endif
}

// ---------------- fp32 -> bf16 streaming convert ----------------
__global__ __launch_bounds__(256) void f32_to_bf16_kernel(
    const float* __restrict__ in, __bf16* __restrict__ out, int n) {
  int i = (blockIdx.x * blockDim.x + threadIdx.x) * 4;
  if (i + 3 < n) {
    float4 v = *(const float4*)(in + i);
    out[i + 0] = (__bf16)v.x;
    out[i + 1] = (__bf16)v.y;
    out[i + 2] = (__bf16)v.z;
    out[i + 3] = (__bf16)v.w;
  } else {
    for (; i < n; ++i) out[i] = (__bf16)in[i];
  }
}

// ---------------- bf16 GEMM: C[M,N] = A[M,K] * W[N,K]^T + bias ----------------
// TDM double-buffered staging.
// mode 0: bf16 [M,N]; 1: bf16 head layout [b,h,q,dh]; 2: fp32 [M,N];
// mode 3: bf16 transposed head layout [b,h,dh,q] (for V)
constexpr int BM = 128, BN = 64, BK = 64, LPAD = 8;
constexpr int LDA_S = BK + LPAD;   // 72 bf16 = 144B rows (16B aligned)
constexpr int LDB_S = BK + LPAD;
// TDM padding codes: 128B rows -> pad_interval 4 (32 DWORDs); 16B pad -> pad_amount 3
constexpr int PAD_IVL64 = 4, PAD_IVL32 = 3, PAD_AMT = 3;

__global__ __launch_bounds__(256) void gemm_bf16_kernel(
    const __bf16* __restrict__ A, const __bf16* __restrict__ W,
    const float* __restrict__ bias, void* __restrict__ outp,
    int M, int N, int K, float scale, int mode) {
  __shared__ __bf16 sA[2][BM * LDA_S];
  __shared__ __bf16 sB[2][BN * LDB_S];

  const int t    = threadIdx.x;
  const int lane = t & 31;
  const int wave = t >> 5;            // 8 waves
  const int wm   = wave >> 1;         // 0..3 : 32 rows each
  const int wn   = wave & 1;          // 0..1 : 32 cols each
  const int l15  = lane & 15;
  const int hf   = lane >> 4;
  const int m0   = blockIdx.y * BM;
  const int n0   = blockIdx.x * BN;

  v8f acc[4] = {};                     // 2x2 tiles of 16x16

  // --- TDM prologue: stream first A/W tiles into buffer 0 ---
  if (wave == 0) {
    tdm_load_2d(lds_byte_off(&sA[0][0]), A + (size_t)m0 * K,
                (unsigned)K, (unsigned)M, BK, BM, (unsigned long long)K,
                PAD_IVL64, PAD_AMT);
    tdm_load_2d(lds_byte_off(&sB[0][0]), W + (size_t)n0 * K,
                (unsigned)K, (unsigned)N, BK, BN, (unsigned long long)K,
                PAD_IVL64, PAD_AMT);
  }

  int cur = 0;
  for (int k0 = 0; k0 < K; k0 += BK) {
    const bool has_next = (k0 + BK < K);
    if (wave == 0) {
      if (has_next) {   // async-issue next tile pair into the other buffer
        tdm_load_2d(lds_byte_off(&sA[cur ^ 1][0]),
                    A + (size_t)m0 * K + (k0 + BK),
                    (unsigned)K, (unsigned)M, BK, BM, (unsigned long long)K,
                    PAD_IVL64, PAD_AMT);
        tdm_load_2d(lds_byte_off(&sB[cur ^ 1][0]),
                    W + (size_t)n0 * K + (k0 + BK),
                    (unsigned)K, (unsigned)N, BK, BN, (unsigned long long)K,
                    PAD_IVL64, PAD_AMT);
        __builtin_amdgcn_s_wait_tensorcnt((short)2);  // current pair complete
      } else {
        __builtin_amdgcn_s_wait_tensorcnt((short)0);
      }
    }
    __syncthreads();   // publish TDM-written LDS to all waves

    // --- 2 K-sub-slabs of 32 : 8 WMMAs per wave per barrier interval ---
#pragma unroll
    for (int ks = 0; ks < 2; ++ks) {
      FragBf fa[2], fb[2];
#pragma unroll
      for (int mi = 0; mi < 2; ++mi) {
        const __bf16* p = &sA[cur][(wm * 32 + mi * 16 + l15) * LDA_S + ks * 32 + hf * 8];
        fa[mi].u[0] = *(const uint4*)p;          // K = kb .. kb+7
        fa[mi].u[1] = *(const uint4*)(p + 16);   // K = kb+16 .. kb+23
      }
#pragma unroll
      for (int ni = 0; ni < 2; ++ni) {
        const __bf16* p = &sB[cur][(wn * 32 + ni * 16 + l15) * LDB_S + ks * 32 + hf * 8];
        fb[ni].u[0] = *(const uint4*)p;          // K = kb .. kb+7
        fb[ni].u[1] = *(const uint4*)(p + 16);   // K = kb+16 .. kb+23
      }
#pragma unroll
      for (int mi = 0; mi < 2; ++mi)
#pragma unroll
        for (int ni = 0; ni < 2; ++ni)
          acc[mi * 2 + ni] = wmma_bf16(fa[mi], fb[ni], acc[mi * 2 + ni]);
    }
    __syncthreads();   // all reads of buf[cur] done before TDM overwrites it
    cur ^= 1;
  }

  // --- epilogue: (acc + bias) * scale ---
#pragma unroll
  for (int mi = 0; mi < 2; ++mi) {
#pragma unroll
    for (int ni = 0; ni < 2; ++ni) {
      v8f c = acc[mi * 2 + ni];
      int gn = n0 + wn * 32 + ni * 16 + l15;
      float bv = bias[gn];
#pragma unroll
      for (int r = 0; r < 8; ++r) {
        int gm = m0 + wm * 32 + mi * 16 + r + hf * 8;
        float val = (c[r] + bv) * scale;
        if (mode == 2) {
          ((float*)outp)[(size_t)gm * N + gn] = val;
        } else if (mode == 1) {
          int bb = gm >> 11;            // / QLEN
          int q  = gm & (QLEN - 1);
          int h  = gn >> 6;             // / DH
          int dh = gn & (DH - 1);
          ((__bf16*)outp)[(((size_t)(bb * NH + h)) * QLEN + q) * DH + dh] = (__bf16)val;
        } else if (mode == 3) {         // transposed head layout [b,h,dh,q]
          int bb = gm >> 11;
          int q  = gm & (QLEN - 1);
          int h  = gn >> 6;
          int dh = gn & (DH - 1);
          ((__bf16*)outp)[(((size_t)(bb * NH + h)) * DH + dh) * QLEN + q] = (__bf16)val;
        } else {
          ((__bf16*)outp)[(size_t)gm * N + gn] = (__bf16)val;
        }
      }
    }
  }
}

// ---------------- flash attention (bf16 WMMA, fp32 softmax, TDM staging) ----------------
constexpr int KC = 32;   // keys per chunk

__global__ __launch_bounds__(256) void attn_kernel(
    const __bf16* __restrict__ Q, const __bf16* __restrict__ Kb,
    const __bf16* __restrict__ Vt, const int* __restrict__ mask,
    __bf16* __restrict__ ctx) {
  __shared__ __bf16 sK[2][KC][DH + 8];     // keys x dh (row major)
  __shared__ __bf16 sVt[2][DH][KC + 8];    // dh x keys (already transposed in global)
  __shared__ __bf16 sP[8][16][KC + 8];     // per-wave P round-trip

  const int t    = threadIdx.x;
  const int lane = t & 31;
  const int wave = t >> 5;
  const int l15  = lane & 15;
  const int hf   = lane >> 4;
  const int b    = blockIdx.z;
  const int h    = blockIdx.y;
  const int q0   = blockIdx.x * 128 + wave * 16;

  const __bf16* Qh  = Q  + ((size_t)(b * NH + h)) * QLEN * DH;  // [q][dh]
  const __bf16* Kh  = Kb + ((size_t)(b * NH + h)) * QLEN * DH;  // [q][dh]
  const __bf16* Vth = Vt + ((size_t)(b * NH + h)) * DH * QLEN;  // [dh][q]

  // Q fragments: 16 rows x 64 dh = two 16x32 A-fragments, loaded once
  FragBf aq[2];
#pragma unroll
  for (int ki = 0; ki < 2; ++ki) {
    const __bf16* p = Qh + (size_t)(q0 + l15) * DH + ki * 32 + hf * 8;
    aq[ki].u[0] = *(const uint4*)p;
    aq[ki].u[1] = *(const uint4*)(p + 16);
  }

  float mrow[8], lrow[8];
  v8f acc[4] = {};
#pragma unroll
  for (int r = 0; r < 8; ++r) { mrow[r] = -3.0e38f; lrow[r] = 0.f; }

  // --- TDM prologue: chunk 0 (K tile + Vt tile) into buffer 0 ---
  if (wave == 0) {
    tdm_load_2d(lds_byte_off(&sK[0][0][0]), Kh,
                DH, QLEN, DH, KC, (unsigned long long)DH, PAD_IVL64, PAD_AMT);
    tdm_load_2d(lds_byte_off(&sVt[0][0][0]), Vth,
                QLEN, DH, KC, DH, (unsigned long long)QLEN, PAD_IVL32, PAD_AMT);
  }

  int cur = 0;
  for (int kc = 0; kc < QLEN; kc += KC) {
    const bool has_next = (kc + KC < QLEN);
    if (wave == 0) {
      if (has_next) {
        tdm_load_2d(lds_byte_off(&sK[cur ^ 1][0][0]), Kh + (size_t)(kc + KC) * DH,
                    DH, QLEN, DH, KC, (unsigned long long)DH, PAD_IVL64, PAD_AMT);
        tdm_load_2d(lds_byte_off(&sVt[cur ^ 1][0][0]), Vth + (kc + KC),
                    QLEN, DH, KC, DH, (unsigned long long)QLEN, PAD_IVL32, PAD_AMT);
        __builtin_amdgcn_s_wait_tensorcnt((short)2);
      } else {
        __builtin_amdgcn_s_wait_tensorcnt((short)0);
      }
    }
    __syncthreads();

    // ---- S = Q * K^T : two 16x16 tiles (keys 0-15 and 16-31) ----
    v8f s0 = {}, s1 = {};
#pragma unroll
    for (int ki = 0; ki < 2; ++ki) {
      FragBf bk0, bk1;
      const __bf16* p0 = &sK[cur][l15][ki * 32 + hf * 8];
      bk0.u[0] = *(const uint4*)p0;  bk0.u[1] = *(const uint4*)(p0 + 16);
      const __bf16* p1 = &sK[cur][16 + l15][ki * 32 + hf * 8];
      bk1.u[0] = *(const uint4*)p1;  bk1.u[1] = *(const uint4*)(p1 + 16);
      s0 = wmma_bf16(aq[ki], bk0, s0);
      s1 = wmma_bf16(aq[ki], bk1, s1);
    }

    // ---- mask + online softmax (row reductions across 16-lane halves) ----
    const float NEG = -3.0e38f;
    const bool kill0 = (mask[b * QLEN + kc + l15] == 0);
    const bool kill1 = (mask[b * QLEN + kc + 16 + l15] == 0);
    float p0a[8], p1a[8], scl[8];
#pragma unroll
    for (int r = 0; r < 8; ++r) {
      float a0 = kill0 ? NEG : s0[r];
      float a1 = kill1 ? NEG : s1[r];
      float v = fmaxf(a0, a1);
#pragma unroll
      for (int off = 1; off < 16; off <<= 1) v = fmaxf(v, __shfl_xor(v, off, 32));
      float mnew = fmaxf(mrow[r], v);
      float sc = __expf(mrow[r] - mnew);
      float e0 = __expf(a0 - mnew);
      float e1 = __expf(a1 - mnew);
      float rs = e0 + e1;
#pragma unroll
      for (int off = 1; off < 16; off <<= 1) rs += __shfl_xor(rs, off, 32);
      lrow[r] = lrow[r] * sc + rs;
      mrow[r] = mnew;
      scl[r] = sc; p0a[r] = e0; p1a[r] = e1;
    }
#pragma unroll
    for (int d = 0; d < 4; ++d)
#pragma unroll
      for (int r = 0; r < 8; ++r) acc[d][r] *= scl[r];

    // ---- P: C-layout -> A-fragment layout via per-wave LDS ----
#pragma unroll
    for (int r = 0; r < 8; ++r) {
      int row = r + hf * 8;
      sP[wave][row][l15]      = (__bf16)p0a[r];
      sP[wave][row][16 + l15] = (__bf16)p1a[r];
    }
    FragBf pf;
    {
      const __bf16* p = &sP[wave][l15][hf * 8];
      pf.u[0] = *(const uint4*)p;
      pf.u[1] = *(const uint4*)(p + 16);
    }

    // ---- acc += P(16x32) * V(32x64) : 4 WMMAs over dh groups ----
#pragma unroll
    for (int d = 0; d < 4; ++d) {
      FragBf vf;
      const __bf16* p = &sVt[cur][d * 16 + l15][hf * 8];
      vf.u[0] = *(const uint4*)p;  vf.u[1] = *(const uint4*)(p + 16);
      acc[d] = wmma_bf16(pf, vf, acc[d]);
    }
    __syncthreads();
    cur ^= 1;
  }

  // ---- normalize and write ctx as [b, q, h*DH + dh] bf16 ----
#pragma unroll
  for (int d = 0; d < 4; ++d) {
    int col = h * DH + d * 16 + l15;
#pragma unroll
    for (int r = 0; r < 8; ++r) {
      int q = q0 + r + hf * 8;
      float inv = 1.0f / lrow[r];
      ctx[((size_t)(b * QLEN + q)) * DIM + col] = (__bf16)(acc[d][r] * inv);
    }
  }
}

// ---------------- host launch ----------------
extern "C" void kernel_launch(void* const* d_in, const int* in_sizes, int n_in,
                              void* d_out, int out_size, void* d_ws, size_t ws_size,
                              hipStream_t stream) {
  (void)in_sizes; (void)n_in; (void)out_size; (void)ws_size;
  const float* x    = (const float*)d_in[0];
  const int*   mask = (const int*)d_in[1];
  const float* wq   = (const float*)d_in[2];
  const float* bq   = (const float*)d_in[3];
  const float* wk   = (const float*)d_in[4];
  const float* bk   = (const float*)d_in[5];
  const float* wv   = (const float*)d_in[6];
  const float* bv   = (const float*)d_in[7];
  const float* wo   = (const float*)d_in[8];
  const float* bo   = (const float*)d_in[9];
  float* out = (float*)d_out;

  // workspace carve-up (bf16 buffers)
  char* ws = (char*)d_ws;
  const size_t SZ_TOK = (size_t)NTOK * DIM * sizeof(__bf16);   // 16 MB
  const size_t SZ_W   = (size_t)DIM * DIM * sizeof(__bf16);    // 2 MB
  __bf16* xb   = (__bf16*)ws;               ws += SZ_TOK;
  __bf16* wqb  = (__bf16*)ws;               ws += SZ_W;
  __bf16* wkb  = (__bf16*)ws;               ws += SZ_W;
  __bf16* wvb  = (__bf16*)ws;               ws += SZ_W;
  __bf16* wob  = (__bf16*)ws;               ws += SZ_W;
  __bf16* qb   = (__bf16*)ws;               ws += SZ_TOK;
  __bf16* kb2  = (__bf16*)ws;               ws += SZ_TOK;
  __bf16* vtb  = (__bf16*)ws;               ws += SZ_TOK;      // V in [b,h,dh,q]
  __bf16* ctxb = (__bf16*)ws;               ws += SZ_TOK;

  // 1) fp32 -> bf16 conversion (x + 4 weight matrices)
  {
    int n = NTOK * DIM;
    f32_to_bf16_kernel<<<(n / 4 + 255) / 256, 256, 0, stream>>>(x, xb, n);
    int nw = DIM * DIM;
    int g = (nw / 4 + 255) / 256;
    f32_to_bf16_kernel<<<g, 256, 0, stream>>>(wq, wqb, nw);
    f32_to_bf16_kernel<<<g, 256, 0, stream>>>(wk, wkb, nw);
    f32_to_bf16_kernel<<<g, 256, 0, stream>>>(wv, wvb, nw);
    f32_to_bf16_kernel<<<g, 256, 0, stream>>>(wo, wob, nw);
  }

  // 2) Q/K/V projections (Q fused with 1/sqrt(DH); V written pre-transposed)
  dim3 gg(DIM / BN, NTOK / BM);   // (16, 64)
  gemm_bf16_kernel<<<gg, 256, 0, stream>>>(xb, wqb, bq, qb,  NTOK, DIM, DIM, QSCALE, 1);
  gemm_bf16_kernel<<<gg, 256, 0, stream>>>(xb, wkb, bk, kb2, NTOK, DIM, DIM, 1.0f,   1);
  gemm_bf16_kernel<<<gg, 256, 0, stream>>>(xb, wvb, bv, vtb, NTOK, DIM, DIM, 1.0f,   3);

  // 3) flash attention (TDM-staged K and V^T)
  dim3 ga(QLEN / 128, NH, BS);    // (16, 16, 4)
  attn_kernel<<<ga, 256, 0, stream>>>(qb, kb2, vtb, mask, ctxb);

  // 4) output projection -> fp32
  gemm_bf16_kernel<<<gg, 256, 0, stream>>>(ctxb, wob, bo, out, NTOK, DIM, DIM, 1.0f, 2);
}
